// MLPEncoder_multi_5540507812006
// MI455X (gfx1250) — matrix operand order repaired and checked by reading, not compiled, not run on hardware
//
#include <hip/hip_runtime.h>
#include <hip/hip_bf16.h>

typedef __bf16 bf16_t;
typedef __bf16 v16bf __attribute__((ext_vector_type(16)));
typedef __bf16 v8bf  __attribute__((ext_vector_type(8)));
typedef float  v8f   __attribute__((ext_vector_type(8)));

#define HB 128   // hidden width
#define NB 64    // nodes
#define EB 4032  // edges
#define BB 64    // batch

// ---------------------------------------------------------------------------
// One 16x128 output strip per wave (8 WMMA accumulators), 8 waves per block.
// D = ELU(A @ W^T + bias), output bf16 [M,128].  K, MODE are compile-time:
//   MODE 0: A is dense [M, K] bf16 (row stride K).
//   MODE 1: K=256, row m = concat(Xn[b,recv[e]], Xn[b,send[e]])      (node2edge)
//   MODE 2: K=384, row m = concat(Xn[b,recv[e]], Xn[b,send[e]], Xe[m]) (edges2)
// Segments are 128 wide, so every 32-wide K-chunk lives in one segment and the
// segment choice folds at compile time after full unroll.
// ---------------------------------------------------------------------------
template <int K, int MODE>
__global__ __launch_bounds__(256) void wmma_gemm_bias_elu(
    const bf16_t* __restrict__ A,
    const bf16_t* __restrict__ W,    // [128, K] row-major (n_out rows)
    const float*  __restrict__ bias, // [128]
    bf16_t* __restrict__ Out,        // [M, 128]
    int M,
    const int* __restrict__ recv, const int* __restrict__ send,
    const bf16_t* __restrict__ Xn, const bf16_t* __restrict__ Xe)
{
    const int lane  = threadIdx.x & 31;
    const int wave  = threadIdx.x >> 5;
    const int mBase = (blockIdx.x * 8 + wave) << 4;
    if (mBase >= M) return;

    const int ml = lane & 15;   // A row within strip / B column within tile
    const int hi = lane >> 4;   // K-half selector (contiguous k = 16*hi + s, same for A and B)
    const int m  = mBase + ml;

    const bf16_t* seg0; const bf16_t* seg1; const bf16_t* seg2;
    if (MODE == 0) {
        seg0 = A + (size_t)m * K; seg1 = seg0; seg2 = seg0;
    } else {
        const int e = m % EB, b = m / EB;
        seg0 = Xn + ((size_t)b * NB + recv[e]) * HB;
        seg1 = Xn + ((size_t)b * NB + send[e]) * HB;
        seg2 = (MODE == 2) ? (Xe + (size_t)m * HB) : seg0;
    }

    // B fragment base: lane's column = t*16 + ml, K contiguous from W row
    const bf16_t* wbase = W + (size_t)ml * K + hi * 16;

    v8f acc[8];
#pragma unroll
    for (int t = 0; t < 8; ++t) acc[t] = (v8f){};

#pragma unroll
    for (int kc = 0; kc < K; kc += 32) {
        const bf16_t* rp;
        int kl;
        if (MODE == 0) { rp = seg0; kl = kc; }
        else {
            const int s = kc >> 7;               // folds after unroll
            rp = (s == 0) ? seg0 : ((s == 1) ? seg1 : seg2);
            kl = kc & (HB - 1);
        }
        // A fragment: lanes<16 K={kl..kl+15}, lanes>=16 K={kl+16..kl+31}
        const bf16_t* ap = rp + kl + hi * 16;
        v8bf a0 = *(const v8bf*)(ap);
        v8bf a1 = *(const v8bf*)(ap + 8);
        v16bf av;
#pragma unroll
        for (int i = 0; i < 8; ++i) { av[i] = a0[i]; av[i + 8] = a1[i]; }

#pragma unroll
        for (int t = 0; t < 8; ++t) {
            v16bf bv = *(const v16bf*)(wbase + (size_t)t * 16 * K + kc);
            acc[t] = __builtin_amdgcn_wmma_f32_16x16x32_bf16(
                false, av, false, bv, (short)0, acc[t], false, false);
        }
    }

    // C/D layout: VGPR r -> row r (lanes<16) / row 8+r (lanes>=16), col = lane%16
#pragma unroll
    for (int t = 0; t < 8; ++t) {
        const int col = t * 16 + ml;
        const float bc = bias[col];
#pragma unroll
        for (int r = 0; r < 8; ++r) {
            const int row = mBase + r + hi * 8;
            float v = acc[t][r] + bc;
            v = v > 0.f ? v : (__expf(v) - 1.f);   // ELU
            Out[(size_t)row * HB + col] = (bf16_t)v;
        }
    }
}

// ---------------------------------------------------------------------------
__global__ void onehot_argmax(const float* __restrict__ rel, int* __restrict__ idx,
                              int E, int N)
{
    int e = blockIdx.x * blockDim.x + threadIdx.x;
    if (e >= E) return;
    int best = 0;
    for (int n = 0; n < N; ++n)
        if (rel[(size_t)e * N + n] > 0.5f) best = n;
    idx[e] = best;
}

__global__ void pad_bf16(const float* __restrict__ src, bf16_t* __restrict__ dst,
                         int rows, int K, int Kp)
{
    size_t i = (size_t)blockIdx.x * blockDim.x + threadIdx.x;
    if (i >= (size_t)rows * Kp) return;
    int r = (int)(i / Kp), k = (int)(i % Kp);
    dst[i] = (bf16_t)(k < K ? src[(size_t)r * K + k] : 0.f);
}

__global__ void zero_f32(float* p, int n)
{
    int i = blockIdx.x * blockDim.x + threadIdx.x;
    if (i < n) p[i] = 0.f;
}

// Column sums / sums-of-squares over [M,128]; two threads per column per block.
__global__ void bn_stats(const bf16_t* __restrict__ X, float* __restrict__ st,
                         int M, int rowsPerBlock)
{
    const int col  = threadIdx.x & 127;
    const int half = threadIdx.x >> 7;
    int r    = blockIdx.x * rowsPerBlock + half;
    int rend = blockIdx.x * rowsPerBlock + rowsPerBlock;
    if (rend > M) rend = M;
    float s = 0.f, q = 0.f;
    for (; r < rend; r += 2) {
        float v = (float)X[(size_t)r * HB + col];
        s += v; q += v * v;
    }
    __shared__ float ls[128], lq[128];
    if (half) { ls[col] = s; lq[col] = q; }
    __syncthreads();
    if (!half) {
        atomicAdd(&st[col],        s + ls[col]);
        atomicAdd(&st[128 + col],  q + lq[col]);
    }
}

__global__ void bn_apply(bf16_t* __restrict__ X, const float* __restrict__ st,
                         const float* __restrict__ gamma, const float* __restrict__ beta,
                         int M)
{
    size_t i = (size_t)blockIdx.x * blockDim.x + threadIdx.x;
    if (i >= (size_t)M * HB) return;
    int col = (int)(i & (HB - 1));
    float mu  = st[col] / (float)M;
    float var = st[128 + col] / (float)M - mu * mu;
    float v = (float)X[i];
    X[i] = (bf16_t)(((v - mu) * rsqrtf(var + 1e-5f)) * gamma[col] + beta[col]);
}

// edge2node: receiver-major edge ordering => 63 consecutive edges per node.
__global__ void edge2node(const bf16_t* __restrict__ X2, bf16_t* __restrict__ Node)
{
    const int bn = blockIdx.x;        // b*64 + n
    const int h  = threadIdx.x;       // 0..127
    const int b  = bn >> 6, n = bn & 63;
    const bf16_t* p = X2 + ((size_t)b * EB + (size_t)n * 63) * HB + h;
    float acc = 0.f;
    for (int j = 0; j < 63; ++j) acc += (float)p[(size_t)j * HB];
    Node[(size_t)bn * HB + h] = (bf16_t)(acc * (1.0f / 64.0f));
}

__global__ void fc_out_kernel(const bf16_t* __restrict__ Y, const float* __restrict__ w,
                              const float* __restrict__ b2, float* __restrict__ out,
                              int typeOff, int M)
{
    int m = blockIdx.x * blockDim.x + threadIdx.x;
    if (m >= M) return;
    const bf16_t* yr = Y + (size_t)m * HB;
    float a0 = 0.f, a1 = 0.f;
    for (int h = 0; h < HB; ++h) {
        float v = (float)yr[h];
        a0 += v * w[h];
        a1 += v * w[HB + h];
    }
    out[(size_t)m * 4 + typeOff + 0] = a0 + b2[0];
    out[(size_t)m * 4 + typeOff + 1] = a1 + b2[1];
}

// ---------------------------------------------------------------------------
extern "C" void kernel_launch(void* const* d_in, const int* in_sizes, int n_in,
                              void* d_out, int out_size, void* d_ws, size_t ws_size,
                              hipStream_t stream)
{
    (void)in_sizes; (void)n_in; (void)out_size; (void)ws_size;
    const float* inputs   = (const float*)d_in[0];
    const float* rel_rec  = (const float*)d_in[1];
    const float* rel_send = (const float*)d_in[2];
    // param index bases: mlp1=3, mlp2=9, mlp3=15, mlp4[0]=21, mlp4[1]=27, fc=33
    auto P = [&](int i) { return (const float*)d_in[i]; };

    const int Mn = BB * NB;   // 4096 node rows
    const int Me = BB * EB;   // 258048 edge rows

    char* ws = (char*)d_ws;
    size_t off = 0;
    auto alloc = [&](size_t bytes) -> char* {
        char* p = ws + off;
        off = (off + bytes + 255) & ~(size_t)255;
        return p;
    };
    int*    recv = (int*)alloc((size_t)EB * 4);
    int*    send = (int*)alloc((size_t)EB * 4);
    bf16_t* W11  = (bf16_t*)alloc((size_t)128 * 224 * 2);
    bf16_t* W21  = (bf16_t*)alloc((size_t)128 * 128 * 2);
    bf16_t* W12  = (bf16_t*)alloc((size_t)128 * 256 * 2);
    bf16_t* W22  = (bf16_t*)alloc((size_t)128 * 128 * 2);
    bf16_t* W13  = (bf16_t*)alloc((size_t)128 * 128 * 2);
    bf16_t* W23  = (bf16_t*)alloc((size_t)128 * 128 * 2);
    bf16_t* W41a = (bf16_t*)alloc((size_t)128 * 384 * 2);
    bf16_t* W42a = (bf16_t*)alloc((size_t)128 * 128 * 2);
    bf16_t* W41b = (bf16_t*)alloc((size_t)128 * 384 * 2);
    bf16_t* W42b = (bf16_t*)alloc((size_t)128 * 128 * 2);
    bf16_t* A0   = (bf16_t*)alloc((size_t)Mn * 224 * 2);
    bf16_t* T1   = (bf16_t*)alloc((size_t)Mn * HB * 2);
    bf16_t* X1   = (bf16_t*)alloc((size_t)Mn * HB * 2);
    bf16_t* NODE = (bf16_t*)alloc((size_t)Mn * HB * 2);
    bf16_t* X3   = (bf16_t*)alloc((size_t)Mn * HB * 2);
    bf16_t* EH   = (bf16_t*)alloc((size_t)Me * HB * 2);
    bf16_t* X2   = (bf16_t*)alloc((size_t)Me * HB * 2);
    bf16_t* Y    = (bf16_t*)alloc((size_t)Me * HB * 2);
    float*  st   = (float*)alloc(256 * 4);
    float*  out  = (float*)d_out;

    auto pad = [&](const float* src, bf16_t* dst, int rows, int K, int Kp) {
        size_t tot = (size_t)rows * Kp;
        pad_bf16<<<(unsigned)((tot + 255) / 256), 256, 0, stream>>>(src, dst, rows, K, Kp);
    };
    auto bn = [&](bf16_t* X, int M, const float* gamma, const float* beta) {
        zero_f32<<<1, 256, 0, stream>>>(st, 256);
        const int rpb = 512;
        bn_stats<<<(M + rpb - 1) / rpb, 256, 0, stream>>>(X, st, M, rpb);
        size_t tot = (size_t)M * HB;
        bn_apply<<<(unsigned)((tot + 255) / 256), 256, 0, stream>>>(X, st, gamma, beta, M);
    };

    // gather indices from one-hot matrices
    onehot_argmax<<<(EB + 255) / 256, 256, 0, stream>>>(rel_rec, recv, EB, NB);
    onehot_argmax<<<(EB + 255) / 256, 256, 0, stream>>>(rel_send, send, EB, NB);

    // bf16 conversions (inputs padded 196 -> 224)
    pad(inputs, A0, Mn, 196, 224);
    pad(P(3),  W11, 128, 196, 224);  pad(P(5),  W21, 128, 128, 128);
    pad(P(9),  W12, 128, 256, 256);  pad(P(11), W22, 128, 128, 128);
    pad(P(15), W13, 128, 128, 128);  pad(P(17), W23, 128, 128, 128);
    pad(P(21), W41a, 128, 384, 384); pad(P(23), W42a, 128, 128, 128);
    pad(P(27), W41b, 128, 384, 384); pad(P(29), W42b, 128, 128, 128);

    // mlp1 on nodes
    wmma_gemm_bias_elu<224, 0><<<Mn / 128, 256, 0, stream>>>(
        A0, W11, P(4), T1, Mn, recv, send, nullptr, nullptr);
    wmma_gemm_bias_elu<128, 0><<<Mn / 128, 256, 0, stream>>>(
        T1, W21, P(6), X1, Mn, recv, send, nullptr, nullptr);
    bn(X1, Mn, P(7), P(8));

    // mlp2 on edges, node2edge gather fused into the GEMM (K=256)
    wmma_gemm_bias_elu<256, 1><<<Me / 128, 256, 0, stream>>>(
        nullptr, W12, P(10), EH, Me, recv, send, X1, nullptr);
    wmma_gemm_bias_elu<128, 0><<<Me / 128, 256, 0, stream>>>(
        EH, W22, P(12), X2, Me, recv, send, nullptr, nullptr);
    bn(X2, Me, P(13), P(14));     // X2 = x_skip

    // edge2node segmented sum / N, then mlp3
    edge2node<<<Mn, HB, 0, stream>>>(X2, NODE);
    wmma_gemm_bias_elu<128, 0><<<Mn / 128, 256, 0, stream>>>(
        NODE, W13, P(16), T1, Mn, recv, send, nullptr, nullptr);
    wmma_gemm_bias_elu<128, 0><<<Mn / 128, 256, 0, stream>>>(
        T1, W23, P(18), X3, Mn, recv, send, nullptr, nullptr);
    bn(X3, Mn, P(19), P(20));

    // per edge-type heads: edges2 gather (recv(x3) | send(x3) | x_skip), K=384
    for (int t = 0; t < 2; ++t) {
        const bf16_t* Wg = t ? W41b : W41a;
        const bf16_t* Wh = t ? W42b : W42a;
        const int base = 21 + 6 * t;
        wmma_gemm_bias_elu<384, 2><<<Me / 128, 256, 0, stream>>>(
            nullptr, Wg, P(base + 1), EH, Me, recv, send, X3, X2);
        wmma_gemm_bias_elu<128, 0><<<Me / 128, 256, 0, stream>>>(
            EH, Wh, P(base + 3), Y, Me, recv, send, nullptr, nullptr);
        bn(Y, Me, P(base + 4), P(base + 5));
        fc_out_kernel<<<(Me + 255) / 256, 256, 0, stream>>>(
            Y, P(33 + 2 * t), P(34 + 2 * t), out, 2 * t, Me);
    }
}
